// SigmoidMoERouter_58652073394405
// MI455X (gfx1250) — matrix operand order repaired and checked by reading, not compiled, not run on hardware
//
#include <hip/hip_runtime.h>
#include <hip/hip_bf16.h>

// ---- problem constants (B=4,S=1024,D=2048,E=16,H=1024,topk=2) ----
#define T_TOK 4096
#define DIM   2048
#define NEXP  16
#define HDIM  1024
#define MTILE 32
#define MAX_TILES 128   // ceil(T_TOK / MTILE)

typedef __attribute__((ext_vector_type(16))) __bf16 v16bf;
typedef __attribute__((ext_vector_type(8)))  float  v8f;

__device__ __forceinline__ v8f zero8() {
  v8f z = {0.f,0.f,0.f,0.f,0.f,0.f,0.f,0.f};
  return z;
}

// ---------- fragment loaders ----------
// A frag (16-bit 16x32): lane row m=lane&15; elems i -> K=kb+i,
// kb = k0 + (lane>>4)*16.
__device__ __forceinline__ v16bf load_a_bf16(const __bf16* row, int kb) {
  union { v16bf v; uint4 q[2]; } u;
  u.q[0] = *(const uint4*)(row + kb);
  u.q[1] = *(const uint4*)(row + kb + 8);
  return u.v;
}
__device__ __forceinline__ v16bf load_a_f32cvt(const float* row, int kb) {
  v16bf v;
  if (row) {
    const float4* p = (const float4*)(row + kb);
    float4 f0 = p[0], f1 = p[1];
    const float4* q = (const float4*)(row + kb + 8);
    float4 g0 = q[0], g1 = q[1];
    v[0]=(__bf16)f0.x; v[1]=(__bf16)f0.y; v[2]=(__bf16)f0.z; v[3]=(__bf16)f0.w;
    v[4]=(__bf16)f1.x; v[5]=(__bf16)f1.y; v[6]=(__bf16)f1.z; v[7]=(__bf16)f1.w;
    v[8]=(__bf16)g0.x; v[9]=(__bf16)g0.y; v[10]=(__bf16)g0.z; v[11]=(__bf16)g0.w;
    v[12]=(__bf16)g1.x; v[13]=(__bf16)g1.y; v[14]=(__bf16)g1.z; v[15]=(__bf16)g1.w;
  } else {
    #pragma unroll
    for (int i = 0; i < 16; ++i) v[i] = (__bf16)0.f;
  }
  return v;
}
// B frag (16-bit 32x16): lane col n=lane&15, elems i -> K=k0+(lane>>4)*16+i.
// Pre-transposed N-major bf16: 16 K-contiguous halves = 32 contiguous bytes.
__device__ __forceinline__ v16bf load_b_bf16(const __bf16* p) {
  union { v16bf v; uint4 q[2]; } u;
  u.q[0] = *(const uint4*)(p);
  u.q[1] = *(const uint4*)(p + 8);
  return u.v;
}
// Fallback: K-major fp32 source, 16 strided loads + cvt.
__device__ __forceinline__ v16bf load_b_f32cvt(const float* wp, int stride) {
  v16bf v;
  #pragma unroll
  for (int i = 0; i < 16; ++i) v[i] = (__bf16)wp[(size_t)i * stride];
  return v;
}

// ---------------- prep: x fp32 -> bf16 ----------------
__global__ void __launch_bounds__(256)
cvt_x(const float* __restrict__ x, __bf16* __restrict__ xb) {
  size_t i = ((size_t)blockIdx.x * 256 + threadIdx.x) * 8;
  float4 f0 = *(const float4*)(x + i);
  float4 f1 = *(const float4*)(x + i + 4);
  union { __bf16 b[8]; uint4 q; } u;
  u.b[0]=(__bf16)f0.x; u.b[1]=(__bf16)f0.y; u.b[2]=(__bf16)f0.z; u.b[3]=(__bf16)f0.w;
  u.b[4]=(__bf16)f1.x; u.b[5]=(__bf16)f1.y; u.b[6]=(__bf16)f1.z; u.b[7]=(__bf16)f1.w;
  *(uint4*)(xb + i) = u.q;
}

// ---------------- prep: W [E][K][N] fp32 -> [E][N][K] bf16 ----------------
__global__ void __launch_bounds__(256)
transpose_cvt(const float* __restrict__ src, __bf16* __restrict__ dst, int K, int N) {
  __shared__ __bf16 tile[32][33];
  const int e = blockIdx.z;
  const float* s = src + (size_t)e * K * N;
  __bf16*      d = dst + (size_t)e * K * N;
  const int n0 = blockIdx.x * 32, k0 = blockIdx.y * 32;
  const int tx = threadIdx.x & 31, ty = threadIdx.x >> 5;  // 32 x 8
  #pragma unroll
  for (int i = 0; i < 4; ++i)
    tile[ty + i * 8][tx] = (__bf16)s[(size_t)(k0 + ty + i * 8) * N + n0 + tx];
  __syncthreads();
  #pragma unroll
  for (int i = 0; i < 4; ++i)
    d[(size_t)(n0 + ty + i * 8) * K + k0 + tx] = tile[tx][ty + i * 8];
}

// ---------------- gate: logits, sigmoid, top-2, routing lists ----------------
__global__ void __launch_bounds__(256)
moe_gate(const float* __restrict__ x, const float* __restrict__ Wg,
         const float* __restrict__ bias,
         int* __restrict__ cnt, int* __restrict__ tokList, float* __restrict__ wList) {
  const int lane = threadIdx.x & 31;
  const int wv   = threadIdx.x >> 5;
  const int t    = blockIdx.x * 8 + wv;
  if (t >= T_TOK) return;
  const float* xr = x + (size_t)t * DIM;

  float acc[NEXP];
  #pragma unroll
  for (int e = 0; e < NEXP; ++e) acc[e] = 0.f;
  for (int d = lane; d < DIM; d += 32) {
    float xv = xr[d];
    #pragma unroll
    for (int e = 0; e < NEXP; ++e) acc[e] += xv * Wg[e * DIM + d];
  }
  #pragma unroll
  for (int e = 0; e < NEXP; ++e) {
    #pragma unroll
    for (int off = 16; off > 0; off >>= 1) acc[e] += __shfl_xor(acc[e], off, 32);
  }
  if (lane == 0) {
    float s[NEXP];
    #pragma unroll
    for (int e = 0; e < NEXP; ++e) s[e] = 1.f / (1.f + __expf(-(acc[e] + bias[e])));
    int i0 = 0; float v0 = s[0];
    #pragma unroll
    for (int e = 1; e < NEXP; ++e) if (s[e] > v0) { v0 = s[e]; i0 = e; }
    int i1 = -1; float v1 = -1.f;
    #pragma unroll
    for (int e = 0; e < NEXP; ++e) if (e != i0 && s[e] > v1) { v1 = s[e]; i1 = e; }
    float inv = 1.f / (v0 + v1 + 1e-6f);
    int p0 = atomicAdd(&cnt[i0], 1);
    tokList[i0 * T_TOK + p0] = t; wList[i0 * T_TOK + p0] = v0 * inv;
    int p1 = atomicAdd(&cnt[i1], 1);
    tokList[i1 * T_TOK + p1] = t; wList[i1 * T_TOK + p1] = v1 * inv;
  }
}

// -------- expert FFN: out[tok] += w * silu(x W1_e) W2_e for a 32-token tile --------
// PRE=true : bf16 staged operands (xb, W1t[e][H][D], W2t[e][D][H])
// PRE=false: raw fp32 operands with inline conversion
template<bool PRE>
__global__ void __launch_bounds__(256, 1)
moe_expert(const float* __restrict__ x,  const __bf16* __restrict__ xb,
           const float* __restrict__ W1, const __bf16* __restrict__ W1t,
           const float* __restrict__ W2, const __bf16* __restrict__ W2t,
           const int* __restrict__ cnt, const int* __restrict__ tokList,
           const float* __restrict__ wList, float* __restrict__ out) {
  const int e    = blockIdx.x;
  const int tile = blockIdx.y;
  const int nTok = cnt[e];
  if (tile * MTILE >= nTok) return;

  __shared__ __align__(16) __bf16 Hs[MTILE][HDIM + 8]; // +8 pad: bank spread, keeps 16B align
  __shared__ int   sTok[MTILE];
  __shared__ float sWt[MTILE];

  const int tid = threadIdx.x;
  if (tid < MTILE) {
    int idx = tile * MTILE + tid;
    if (idx < nTok) { sTok[tid] = tokList[e * T_TOK + idx]; sWt[tid] = wList[e * T_TOK + idx]; }
    else            { sTok[tid] = -1;                       sWt[tid] = 0.f; }
  }
  __syncthreads();

  const int lane  = tid & 31;
  const int wv    = tid >> 5;
  const int lhalf = lane >> 4;
  const int l16   = lane & 15;

  // ---------------- Phase A: Hs = bf16( silu(X @ W1e) * w ) ----------------
  {
    const int tok0 = sTok[l16];
    const int tok1 = sTok[16 + l16];
    const float*  xr0 = tok0 >= 0 ? x  + (size_t)tok0 * DIM : nullptr;
    const float*  xr1 = tok1 >= 0 ? x  + (size_t)tok1 * DIM : nullptr;
    const __bf16* xb0 = xb + (size_t)(tok0 >= 0 ? tok0 : 0) * DIM;  // tok<0 rows zeroed by wt=0 & silu(garbage)*0? -> must be zero; guard below
    const __bf16* xb1 = xb + (size_t)(tok1 >= 0 ? tok1 : 0) * DIM;

    for (int pass = 0; pass < 2; ++pass) {            // 2 x 64 cols per wave
      const int nbase = wv * 128 + pass * 64;
      v8f acc[2][4];
      #pragma unroll
      for (int mt = 0; mt < 2; ++mt)
        #pragma unroll
        for (int nt = 0; nt < 4; ++nt) acc[mt][nt] = zero8();

      for (int k0 = 0; k0 < DIM; k0 += 32) {
        const int kb = k0 + lhalf * 16;
        v16bf a0, a1;
        if (PRE) { a0 = load_a_bf16(xb0, kb); a1 = load_a_bf16(xb1, kb); }
        else     { a0 = load_a_f32cvt(xr0, kb); a1 = load_a_f32cvt(xr1, kb); }
        #pragma unroll
        for (int nt = 0; nt < 4; ++nt) {
          const int n0 = nbase + nt * 16;
          v16bf b;
          if (PRE)  // W1t[e][n][k], K-contiguous
            b = load_b_bf16(W1t + (size_t)e * DIM * HDIM + (size_t)(n0 + l16) * DIM
                                + k0 + lhalf * 16);
          else
            b = load_b_f32cvt(W1 + (size_t)e * DIM * HDIM
                                 + (size_t)(k0 + lhalf * 16) * HDIM + n0 + l16, HDIM);
          acc[0][nt] = __builtin_amdgcn_wmma_f32_16x16x32_bf16(false, a0, false, b,
                         (short)0, acc[0][nt], false, false);
          acc[1][nt] = __builtin_amdgcn_wmma_f32_16x16x32_bf16(false, a1, false, b,
                         (short)0, acc[1][nt], false, false);
        }
      }
      #pragma unroll
      for (int mt = 0; mt < 2; ++mt) {
        #pragma unroll
        for (int nt = 0; nt < 4; ++nt) {
          const int ncol = nbase + nt * 16 + l16;
          #pragma unroll
          for (int r = 0; r < 8; ++r) {
            const int m = mt * 16 + lhalf * 8 + r;
            float hv = acc[mt][nt][r];
            float sv = hv / (1.f + __expf(-hv));       // silu
            Hs[m][ncol] = (__bf16)(sv * sWt[m]);       // wt==0 for pad rows
          }
        }
      }
    }
  }
  __syncthreads();

  // ---------------- Phase B: out += Hs @ W2e ----------------
  for (int pass = 0; pass < 4; ++pass) {               // 4 x 64 cols per wave
    const int nbase = wv * 256 + pass * 64;
    v8f acc[2][4];
    #pragma unroll
    for (int mt = 0; mt < 2; ++mt)
      #pragma unroll
      for (int nt = 0; nt < 4; ++nt) acc[mt][nt] = zero8();

    for (int k0 = 0; k0 < HDIM; k0 += 32) {
      const int kb = k0 + lhalf * 16;
      v16bf a0 = load_a_bf16(&Hs[l16][0],      kb);
      v16bf a1 = load_a_bf16(&Hs[16 + l16][0], kb);
      #pragma unroll
      for (int nt = 0; nt < 4; ++nt) {
        const int n0 = nbase + nt * 16;
        v16bf b;
        if (PRE)  // W2t[e][n][k], K-contiguous
          b = load_b_bf16(W2t + (size_t)e * HDIM * DIM + (size_t)(n0 + l16) * HDIM
                              + k0 + lhalf * 16);
        else
          b = load_b_f32cvt(W2 + (size_t)e * HDIM * DIM
                               + (size_t)(k0 + lhalf * 16) * DIM + n0 + l16, DIM);
        acc[0][nt] = __builtin_amdgcn_wmma_f32_16x16x32_bf16(false, a0, false, b,
                       (short)0, acc[0][nt], false, false);
        acc[1][nt] = __builtin_amdgcn_wmma_f32_16x16x32_bf16(false, a1, false, b,
                       (short)0, acc[1][nt], false, false);
      }
    }
    #pragma unroll
    for (int mt = 0; mt < 2; ++mt) {
      #pragma unroll
      for (int nt = 0; nt < 4; ++nt) {
        const int ncol = nbase + nt * 16 + l16;
        #pragma unroll
        for (int r = 0; r < 8; ++r) {
          const int m = mt * 16 + lhalf * 8 + r;
          const int tok = sTok[m];
          if (tok >= 0)
            unsafeAtomicAdd(out + (size_t)tok * DIM + ncol, acc[mt][nt][r]);
        }
      }
    }
  }
}

extern "C" void kernel_launch(void* const* d_in, const int* in_sizes, int n_in,
                              void* d_out, int out_size, void* d_ws, size_t ws_size,
                              hipStream_t stream) {
  const float* x    = (const float*)d_in[0];   // [4,1024,2048]
  const float* Wg   = (const float*)d_in[1];   // [16,2048]
  const float* bias = (const float*)d_in[2];   // [16]
  const float* W1   = (const float*)d_in[3];   // [16,2048,1024]
  const float* W2   = (const float*)d_in[4];   // [16,1024,2048]
  float* out = (float*)d_out;                  // [4,1024,2048]

  char* w = (char*)d_ws;
  int*   cnt     = (int*)w;                                         // 64B (reserve 256)
  int*   tokList = (int*)(w + 256);                                 // 256KB
  float* wList   = (float*)(w + 256 + (size_t)NEXP * T_TOK * 4);    // 256KB
  size_t off = 256 + 2ull * NEXP * T_TOK * 4;
  __bf16* xb  = (__bf16*)(w + off);                                 // 16.8MB
  __bf16* W1t = xb  + (size_t)T_TOK * DIM;                          // 67MB
  __bf16* W2t = W1t + (size_t)NEXP * DIM * HDIM;                    // 67MB
  const size_t need = off + 2ull * ((size_t)T_TOK * DIM
                        + 2ull * NEXP * DIM * HDIM);                // ~151.5MB

  hipMemsetAsync(cnt, 0, NEXP * sizeof(int), stream);
  hipMemsetAsync(out, 0, (size_t)T_TOK * DIM * sizeof(float), stream);

  moe_gate<<<T_TOK / 8, 256, 0, stream>>>(x, Wg, bias, cnt, tokList, wList);

  if (ws_size >= need) {
    cvt_x<<<(T_TOK * DIM) / (256 * 8), 256, 0, stream>>>(x, xb);
    transpose_cvt<<<dim3(HDIM / 32, DIM / 32, NEXP), 256, 0, stream>>>(W1, W1t, DIM, HDIM);
    transpose_cvt<<<dim3(DIM / 32, HDIM / 32, NEXP), 256, 0, stream>>>(W2, W2t, HDIM, DIM);
    moe_expert<true><<<dim3(NEXP, MAX_TILES), 256, 0, stream>>>(
        x, xb, W1, W1t, W2, W2t, cnt, tokList, wList, out);
  } else {
    moe_expert<false><<<dim3(NEXP, MAX_TILES), 256, 0, stream>>>(
        x, xb, W1, W1t, W2, W2t, cnt, tokList, wList, out);
  }
}
